// AttentiveBP_52209622450254
// MI455X (gfx1250) — matrix operand order repaired and checked by reading, not compiled, not run on hardware
//
#include <hip/hip_runtime.h>
#include <hip/hip_bf16.h>
#include <math.h>

// ---------------- problem constants (match reference setup_inputs) -------
#define F_N   100000
#define V_N   30000
#define D_DIM 15
#define HID   60
#define T_N   (2 * F_N)

typedef __attribute__((ext_vector_type(16))) _Float16 v16h;
typedef __attribute__((ext_vector_type(8)))  float    v8f;

static __device__ __forceinline__ float sigmoidf_(float x) {
  return 1.0f / (1.0f + __expf(-x));
}

// ---------------- zero init ----------------------------------------------
__global__ void k_zero(float* __restrict__ belief, float* __restrict__ accum, int n) {
  int i = blockIdx.x * blockDim.x + threadIdx.x;
  if (i < n) belief[i] = 0.0f;
  if (i < 4) accum[i] = 0.0f;
}

// ---------------- min-plus BP factor update ------------------------------
// m_f2rv[f,i] = min_j cost[f,i,j] + m_cv2f[f,j]
// m_f2cv[f,j] = min_i cost[f,i,j] + m_rv2f[f,i]
__global__ void k_minplus(const float* __restrict__ msgs, const float* __restrict__ cost,
                          const int* __restrict__ rv2f_i, const int* __restrict__ cv2f_i,
                          const int* __restrict__ f2rv_i, const int* __restrict__ f2cv_i,
                          float* __restrict__ msgs_work) {
  int f = blockIdx.x * blockDim.x + threadIdx.x;
  if (f >= F_N) return;
  float rv[D_DIM], cv[D_DIM], fr[D_DIM], fc[D_DIM];
  const float* rvp = msgs + (size_t)rv2f_i[f] * D_DIM;
  const float* cvp = msgs + (size_t)cv2f_i[f] * D_DIM;
#pragma unroll
  for (int j = 0; j < D_DIM; ++j) { rv[j] = rvp[j]; cv[j] = cvp[j]; fc[j] = 3.0e38f; }
  const float* cp = cost + (size_t)f * (D_DIM * D_DIM);
#pragma unroll
  for (int i = 0; i < D_DIM; ++i) {
    float rmin = 3.0e38f;
    float ri = rv[i];
#pragma unroll
    for (int j = 0; j < D_DIM; ++j) {
      float c = cp[i * D_DIM + j];
      rmin  = fminf(rmin, c + cv[j]);
      fc[j] = fminf(fc[j], c + ri);
    }
    fr[i] = rmin;
  }
  float* o1 = msgs_work + (size_t)f2rv_i[f] * D_DIM;
  float* o2 = msgs_work + (size_t)f2cv_i[f] * D_DIM;
#pragma unroll
  for (int j = 0; j < D_DIM; ++j) { o1[j] = fr[j]; o2[j] = fc[j]; }
}

// ---------------- belief scatter -----------------------------------------
__global__ void k_belief(const float* __restrict__ msgs_work, const int* __restrict__ rows,
                         const int* __restrict__ segs, float* __restrict__ belief) {
  int i = blockIdx.x * blockDim.x + threadIdx.x;
  if (i >= T_N * D_DIM) return;
  int t = i / D_DIM;
  int j = i - t * D_DIM;
  atomicAdd(&belief[(size_t)segs[t] * D_DIM + j],
            msgs_work[(size_t)rows[t] * D_DIM + j]);
}

// ---------------- softmax(-belief), entropy ------------------------------
__global__ void k_dist(const float* __restrict__ belief, float* __restrict__ dist,
                       float* __restrict__ accum) {
  __shared__ float s_ent;
  if (threadIdx.x == 0) s_ent = 0.0f;
  __syncthreads();
  int v = blockIdx.x * blockDim.x + threadIdx.x;
  float ent = 0.0f;
  if (v < V_N) {
    float b[D_DIM];
    float bmin = 3.0e38f;
#pragma unroll
    for (int j = 0; j < D_DIM; ++j) { b[j] = belief[(size_t)v * D_DIM + j]; bmin = fminf(bmin, b[j]); }
    float e[D_DIM];
    float s = 0.0f;
#pragma unroll
    for (int j = 0; j < D_DIM; ++j) { e[j] = __expf(bmin - b[j]); s += e[j]; }
    float inv = 1.0f / s;
#pragma unroll
    for (int j = 0; j < D_DIM; ++j) {
      float dj = e[j] * inv;
      dist[(size_t)v * D_DIM + j] = dj;
      ent -= dj * log2f(dj + 1e-6f);
    }
  }
  atomicAdd(&s_ent, ent);
  __syncthreads();
  if (threadIdx.x == 0) atomicAdd(&accum[0], s_ent);
}

// ---------------- per-factor bilinear form + argmax cost -----------------
__global__ void k_factor(const float* __restrict__ cost, const float* __restrict__ dist,
                         const int* __restrict__ rv_idx, const int* __restrict__ cv_idx,
                         float* __restrict__ accum) {
  __shared__ float s_loss, s_cost;
  if (threadIdx.x == 0) { s_loss = 0.0f; s_cost = 0.0f; }
  __syncthreads();
  int f = blockIdx.x * blockDim.x + threadIdx.x;
  float pf = 0.0f, cval = 0.0f;
  if (f < F_N) {
    float dr[D_DIM], dc[D_DIM];
    const float* drp = dist + (size_t)rv_idx[f] * D_DIM;
    const float* dcp = dist + (size_t)cv_idx[f] * D_DIM;
    int vr = 0, vc = 0;
    float mr = -1.0f, mc = -1.0f;
#pragma unroll
    for (int j = 0; j < D_DIM; ++j) {
      dr[j] = drp[j]; dc[j] = dcp[j];
      if (dr[j] > mr) { mr = dr[j]; vr = j; }   // first-max == jnp.argmax
      if (dc[j] > mc) { mc = dc[j]; vc = j; }
    }
    const float* cp = cost + (size_t)f * (D_DIM * D_DIM);
#pragma unroll
    for (int i = 0; i < D_DIM; ++i) {
      float acc = 0.0f;
#pragma unroll
      for (int j = 0; j < D_DIM; ++j) acc += cp[i * D_DIM + j] * dc[j];
      pf += dr[i] * acc;
    }
    cval = cp[vr * D_DIM + vc];
  }
  atomicAdd(&s_loss, pf);
  atomicAdd(&s_cost, cval);
  __syncthreads();
  if (threadIdx.x == 0) {
    atomicAdd(&accum[1], s_loss);
    atomicAdd(&accum[2], s_cost);
  }
}

// ---------------- finalize two scalars -----------------------------------
__global__ void k_final(const float* __restrict__ accum, float* __restrict__ out) {
  if (blockIdx.x == 0 && threadIdx.x == 0) {
    out[0] = accum[1] + 0.1f * (accum[0] / (float)V_N);  // loss
    out[1] = accum[2];                                    // cost_mean
  }
}

// ====================== WMMA GRU (reference GEMM work) ====================
// gi = x @ w_ih^T, gh = h @ w_hh^T  (180 outputs = 3 gates x 60)
// Weights staged once per block into zero-padded f16 LDS (192 rows):
//   s_wih: [192][16]  (K padded 15->16)
//   s_whh: [192][64]  (K padded 60->64)
// A layout (16-bit A 16x32): lanes 0-15 M=lane hold K kc+0..15;
// lanes 16-31 M=lane-16 hold K kc+16..31 (matches B).  KC/KVALID are template
// constants so per-element validity folds at compile time -> contiguous loads.
template <int KC, int KVALID>
__device__ __forceinline__ v16h load_a16(const float* __restrict__ rp, int lane) {
  v16h a;
  if (lane < 16) {
#pragma unroll
    for (int e = 0; e < 16; ++e) {
      const int k = KC + e;        // compile-time
      a[e] = (k < KVALID) ? (_Float16)rp[k] : (_Float16)0.0f;
    }
  } else {
#pragma unroll
    for (int e = 0; e < 16; ++e) {
      const int k = KC + 16 + e;    // compile-time
      a[e] = (k < KVALID) ? (_Float16)rp[k] : (_Float16)0.0f;
    }
  }
  return a;
}

// B layout (16-bit B 32x16): lane n (0-15) holds column N=n, K=kc+0..15;
// lanes 16-31 hold column N=lane-16, K=kc+16..31.  B[k,n] = Wlds[n][k],
// zero-padded, so 16 consecutive halves (32B) per lane, no guards.
__device__ __forceinline__ v16h load_b_lds(const _Float16* __restrict__ lds,
                                           int nbase, int kc, int ldk, int lane) {
  const _Float16* p = lds + (nbase + (lane & 15)) * ldk + kc + ((lane >= 16) ? 16 : 0);
  v16h b;
#pragma unroll
  for (int e = 0; e < 16; ++e) b[e] = p[e];
  return b;
}

__global__ void __launch_bounds__(128)
k_gru_wmma(const float* __restrict__ xsrc, const int* __restrict__ xidxA,
           const int* __restrict__ xidxB, const float* __restrict__ h,
           const float* __restrict__ w_ih, const float* __restrict__ w_hh,
           const float* __restrict__ b_ih, const float* __restrict__ b_hh,
           float* __restrict__ hout) {
  __shared__ _Float16 s_wih[192 * 16];
  __shared__ _Float16 s_whh[192 * 64];

  // cooperative zero-padded f16 staging of both weight matrices
  for (int i = threadIdx.x; i < 192 * 16; i += blockDim.x) {
    int n = i >> 4, k = i & 15;
    float v = (n < 180 && k < D_DIM) ? w_ih[n * D_DIM + k] : 0.0f;
    s_wih[i] = (_Float16)v;
  }
  for (int i = threadIdx.x; i < 192 * 64; i += blockDim.x) {
    int n = i >> 6, k = i & 63;
    float v = (n < 180 && k < HID) ? w_hh[n * HID + k] : 0.0f;
    s_whh[i] = (_Float16)v;
  }
  __syncthreads();

  // grid is exact: (T_N/16) M-tiles * 4 N-subtiles waves, 4 waves/block
  int gw   = (blockIdx.x * blockDim.x + threadIdx.x) >> 5;  // global wave id
  int lane = threadIdx.x & 31;
  int mtile = gw >> 2, usub = gw & 3;
  int m0 = mtile * 16, u0 = usub * 16;

  v8f ir = {}, iz = {}, in_ = {}, hr = {}, hz = {}, hn = {};

  {  // input GEMM: K = 15 (one padded K-chunk), gathered rows
    int mrow = m0 + (lane & 15);
    int row  = (mrow < F_N) ? xidxA[mrow] : xidxB[mrow - F_N];
    v16h a  = load_a16<0, D_DIM>(xsrc + (size_t)row * D_DIM, lane);
    v16h b0 = load_b_lds(s_wih,   0 + u0, 0, 16, lane);
    ir  = __builtin_amdgcn_wmma_f32_16x16x32_f16(false, a, false, b0, (short)0, ir,  false, false);
    v16h b1 = load_b_lds(s_wih,  60 + u0, 0, 16, lane);
    iz  = __builtin_amdgcn_wmma_f32_16x16x32_f16(false, a, false, b1, (short)0, iz,  false, false);
    v16h b2 = load_b_lds(s_wih, 120 + u0, 0, 16, lane);
    in_ = __builtin_amdgcn_wmma_f32_16x16x32_f16(false, a, false, b2, (short)0, in_, false, false);
  }
  {  // hidden GEMM: K = 60, chunk kc=0 (fully valid)
    const float* rp = h + (size_t)(m0 + (lane & 15)) * HID;
    v16h a  = load_a16<0, HID>(rp, lane);
    v16h b0 = load_b_lds(s_whh,   0 + u0, 0, 64, lane);
    hr = __builtin_amdgcn_wmma_f32_16x16x32_f16(false, a, false, b0, (short)0, hr, false, false);
    v16h b1 = load_b_lds(s_whh,  60 + u0, 0, 64, lane);
    hz = __builtin_amdgcn_wmma_f32_16x16x32_f16(false, a, false, b1, (short)0, hz, false, false);
    v16h b2 = load_b_lds(s_whh, 120 + u0, 0, 64, lane);
    hn = __builtin_amdgcn_wmma_f32_16x16x32_f16(false, a, false, b2, (short)0, hn, false, false);
  }
  {  // hidden GEMM: chunk kc=32 (K 32..59 valid, 60..63 zero at compile time)
    const float* rp = h + (size_t)(m0 + (lane & 15)) * HID;
    v16h a  = load_a16<32, HID>(rp, lane);
    v16h b0 = load_b_lds(s_whh,   0 + u0, 32, 64, lane);
    hr = __builtin_amdgcn_wmma_f32_16x16x32_f16(false, a, false, b0, (short)0, hr, false, false);
    v16h b1 = load_b_lds(s_whh,  60 + u0, 32, 64, lane);
    hz = __builtin_amdgcn_wmma_f32_16x16x32_f16(false, a, false, b1, (short)0, hz, false, false);
    v16h b2 = load_b_lds(s_whh, 120 + u0, 32, 64, lane);
    hn = __builtin_amdgcn_wmma_f32_16x16x32_f16(false, a, false, b2, (short)0, hn, false, false);
  }

  // C/D layout: VGPR r -> M = r (+8 for lanes 16-31), N = lane&15
  int u = u0 + (lane & 15);
  int rowadd = (lane >= 16) ? 8 : 0;
  if (u < HID) {
    float bir = b_ih[u], biz = b_ih[60 + u], bin = b_ih[120 + u];
    float bhr = b_hh[u], bhz = b_hh[60 + u], bhn = b_hh[120 + u];
#pragma unroll
    for (int r = 0; r < 8; ++r) {
      int t = m0 + r + rowadd;
      float rg = sigmoidf_(ir[r] + bir + hr[r] + bhr);
      float zg = sigmoidf_(iz[r] + biz + hz[r] + bhz);
      float hv = h[(size_t)t * HID + u];
      float ng = tanhf(in_[r] + bin + rg * (hn[r] + bhn));
      hout[(size_t)t * HID + u] = (1.0f - zg) * ng + zg * hv;
    }
  }
}

// ========================== host launcher ================================
extern "C" void kernel_launch(void* const* d_in, const int* in_sizes, int n_in,
                              void* d_out, int out_size, void* d_ws, size_t ws_size,
                              hipStream_t stream) {
  const float* msgs       = (const float*)d_in[0];
  const float* cost       = (const float*)d_in[1];
  const float* v2f_hidden = (const float*)d_in[3];
  const float* f2v_hidden = (const float*)d_in[4];
  const float* w_ih1 = (const float*)d_in[6];
  const float* w_hh1 = (const float*)d_in[7];
  const float* b_ih1 = (const float*)d_in[8];
  const float* b_hh1 = (const float*)d_in[9];
  const float* w_ih2 = (const float*)d_in[10];
  const float* w_hh2 = (const float*)d_in[11];
  const float* b_ih2 = (const float*)d_in[12];
  const float* b_hh2 = (const float*)d_in[13];
  const int* rv2f_i   = (const int*)d_in[36];
  const int* cv2f_i   = (const int*)d_in[37];
  const int* f2rv_i   = (const int*)d_in[38];
  const int* f2cv_i   = (const int*)d_in[39];
  const int* f2v_rows = (const int*)d_in[42];
  const int* f2v_segs = (const int*)d_in[46];
  const int* rv_idx   = (const int*)d_in[48];
  const int* cv_idx   = (const int*)d_in[49];

  // workspace carve-out
  char* ws = (char*)d_ws;
  size_t off = 0;
  auto carve = [&](size_t bytes) -> void* {
    void* p = (void*)(ws + off);
    off = (off + bytes + 255) & ~(size_t)255;
    return p;
  };
  float* msgs_work = (float*)carve((size_t)4 * F_N * D_DIM * sizeof(float)); // 24 MB
  float* belief    = (float*)carve((size_t)V_N * D_DIM * sizeof(float));
  float* dist      = (float*)carve((size_t)V_N * D_DIM * sizeof(float));
  float* accum     = (float*)carve(256);
  size_t live_off  = off;
  float* gru_out   = (float*)carve((size_t)T_N * HID * sizeof(float));       // 48 MB
  size_t full_off  = off;
  (void)in_sizes; (void)n_in; (void)out_size;

  // ---- live path (fully determines d_out) ----
  hipMemcpyAsync(msgs_work, msgs, (size_t)4 * F_N * D_DIM * sizeof(float),
                 hipMemcpyDeviceToDevice, stream);
  int nz = V_N * D_DIM;
  k_zero<<<(nz + 255) / 256, 256, 0, stream>>>(belief, accum, nz);
  k_minplus<<<(F_N + 127) / 128, 128, 0, stream>>>(msgs, cost, rv2f_i, cv2f_i,
                                                   f2rv_i, f2cv_i, msgs_work);
  int nb = T_N * D_DIM;
  k_belief<<<(nb + 255) / 256, 256, 0, stream>>>(msgs_work, f2v_rows, f2v_segs, belief);
  k_dist<<<(V_N + 255) / 256, 256, 0, stream>>>(belief, dist, accum);
  k_factor<<<(F_N + 255) / 256, 256, 0, stream>>>(cost, dist, rv_idx, cv_idx, accum);
  k_final<<<1, 32, 0, stream>>>(accum, (float*)d_out);

  // ---- GRU hidden-state updates via v_wmma (reference work; output-dead) ----
  if (full_off <= ws_size && live_off <= ws_size) {
    int nblocks = T_N / 16;   // 4 waves/block, one wave per (M-tile, N-subtile)
    k_gru_wmma<<<nblocks, 128, 0, stream>>>(
        msgs, rv2f_i, cv2f_i, v2f_hidden, w_ih1, w_hh1, b_ih1, b_hh1, gru_out);
    k_gru_wmma<<<nblocks, 128, 0, stream>>>(
        msgs_work, f2rv_i, f2cv_i, f2v_hidden, w_ih2, w_hh2, b_ih2, b_hh2, gru_out);
  }
}
